// _DialogueGCNLayer_1073741824706
// MI455X (gfx1250) — matrix operand order repaired and checked by reading, not compiled, not run on hardware
//
#include <hip/hip_runtime.h>
#include <cstdint>

// Problem constants (match reference): B=8, R=5, N=1024, D=512
#define BB 8
#define RR 5
#define NN 1024
#define DD 512
#define LN_EPS 1e-5f

// ------------------------- WMMA types (gfx1250, wave32) -------------------------
typedef __attribute__((ext_vector_type(16))) __bf16    v16bf;
typedef __attribute__((ext_vector_type(8)))  float     v8f;
typedef __attribute__((ext_vector_type(8)))  uint32_t  v8u;

// LDS tile strides (dwords). Padded to stay 16B-aligned and spread banks.
#define LDA 20   // 32 bf16 = 16 dw payload + 4 dw pad  (A tile rows)
#define LDB 36   // 64 bf16 = 32 dw payload + 4 dw pad  (B tile rows)

// ------------------------- CDNA5 async global->LDS copy -------------------------
#if defined(__has_builtin)
#  if __has_builtin(__builtin_amdgcn_global_load_async_to_lds_b128)
#    define HAVE_ASYNC_LDS 1
#  endif
#  if __has_builtin(__builtin_amdgcn_s_wait_asynccnt)
#    define HAVE_WAIT_ASYNC 1
#  endif
#endif

// Builtin signature (from hipcc diagnostic): pointee is GCC-vector int4,
// global operand in target AS1, LDS operand in target AS3.
typedef int v4i_t __attribute__((vector_size(16)));
typedef __attribute__((address_space(1))) v4i_t as1_v4i;
typedef __attribute__((address_space(3))) v4i_t as3_v4i;

template <int N>
__device__ __forceinline__ void wait_asynccnt() {
#if defined(HAVE_WAIT_ASYNC)
  __builtin_amdgcn_s_wait_asynccnt(N);
#else
  asm volatile("s_wait_asynccnt %0" ::"i"(N) : "memory");
#endif
}

// 16B async DMA: global -> LDS, tracked by ASYNCcnt (no VGPR round-trip).
__device__ __forceinline__ void async_cp16(void* lds, const void* g) {
#if defined(HAVE_ASYNC_LDS)
  __builtin_amdgcn_global_load_async_to_lds_b128(
      (as1_v4i*)(uintptr_t)g,
      (as3_v4i*)(uint32_t)(uintptr_t)lds,  // low 32 bits == LDS offset
      0, 0);
#else
  // VDST = LDS address VGPR, VADDR = 64-bit global address (ISA 15.18.3 op 98)
  asm volatile("global_load_async_to_lds_b128 %0, %1, off" ::"v"(
                   (uint32_t)(uintptr_t)lds),
               "v"((uint64_t)(uintptr_t)g)
               : "memory");
#endif
}

// ------------------------- helpers -------------------------
__device__ __forceinline__ uint32_t f2bf1(float a) {
  uint32_t u = __float_as_uint(a);
  return (u + 0x7FFFu + ((u >> 16) & 1u)) >> 16;   // round-to-nearest-even
}
__device__ __forceinline__ uint32_t pack2bf(float a, float b) {
  return (f2bf1(b) << 16) | (f2bf1(a) & 0xFFFFu);
}

// A fragment (16x32 bf16): lane l -> row M=l&15; K chunks at dw (l>>4)*4 and +8.
__device__ __forceinline__ v16bf fragA(const uint32_t* p) {
  uint4 lo = *reinterpret_cast<const uint4*>(p);
  uint4 hi = *reinterpret_cast<const uint4*>(p + 8);
  v8u u; u[0]=lo.x; u[1]=lo.y; u[2]=lo.z; u[3]=lo.w;
         u[4]=hi.x; u[5]=hi.y; u[6]=hi.z; u[7]=hi.w;
  return __builtin_bit_cast(v16bf, u);
}
// B fragment (32x16 bf16): lane l -> column N=col; K rows (l>>4)*8 + {0..7, 16..23}.
__device__ __forceinline__ v16bf fragB(const uint32_t* p, int col) {
  const uint16_t* q = reinterpret_cast<const uint16_t*>(p);
  uint16_t t[16];
#pragma unroll
  for (int s = 0; s < 16; ++s) t[s] = q[(s < 8 ? s : s + 8) * 2 * LDB + col];
  return __builtin_bit_cast(v16bf, t);
}

// ------------------------- 1) degree reciprocal -------------------------
__global__ __launch_bounds__(256) void k_deg(const float* __restrict__ adj,
                                             float* __restrict__ rdeg) {
  __shared__ float red[256];
  const int row = blockIdx.x;                 // b*R*N + r*N + v
  const int tid = threadIdx.x;
  const float* p = adj + (size_t)row * NN;
  float s = p[tid] + p[tid + 256] + p[tid + 512] + p[tid + 768];
  red[tid] = s;
  __syncthreads();
  for (int off = 128; off > 0; off >>= 1) {
    if (tid < off) red[tid] += red[tid + off];
    __syncthreads();
  }
  if (tid == 0) rdeg[row] = 1.0f / fmaxf(red[0], 1.0f);
}

// ------------------------- 2) h -> bf16 (packed pairs) -------------------------
__global__ __launch_bounds__(256) void k_cvt_h(const float* __restrict__ h,
                                               uint32_t* __restrict__ hbf) {
  const int i = blockIdx.x * 256 + threadIdx.x;   // dword index
  hbf[i] = pack2bf(h[2 * i], h[2 * i + 1]);
}

// ------------------------- 3) W -> bf16, transposed to (r, d, e) --------------
__global__ __launch_bounds__(256) void k_cvt_wt(const float* __restrict__ W,
                                                uint16_t* __restrict__ wt) {
  const int i = blockIdx.x * 256 + threadIdx.x;   // < R*D*D, layout (r,d,e)
  const int r = i / (DD * DD);
  const int rem = i - r * DD * DD;
  const int d = rem >> 9;
  const int e = rem & (DD - 1);
  wt[i] = (uint16_t)f2bf1(W[((size_t)r * DD + e) * DD + d]);  // W is (r,e,d)
}

// ------------------------- 4) hW[b,r] = h_bf @ Wt_r  (M=1024,N=512,K=512) -----
// 3-deep async LDS pipeline, 1 barrier per K-step. Both tiles are pure copies.
__global__ __launch_bounds__(256) void k_gemm_hw(const uint32_t* __restrict__ hbf,
                                                 const uint32_t* __restrict__ wt,
                                                 uint16_t* __restrict__ hw) {
  __shared__ __align__(16) uint32_t sA[3][128 * LDA];
  __shared__ __align__(16) uint32_t sB[3][32 * LDB];
  const int e0 = blockIdx.x * 64;
  const int u0 = blockIdx.y * 128;
  const int br = blockIdx.z;                 // b*R + r
  const int b  = br / RR;
  const int r  = br - b * RR;
  const int tid = threadIdx.x, lane = tid & 31, wv = tid >> 5;

  // per-thread async copies for one K-tile: 2x16B into A, 1x16B into B
  auto issueTile = [&](int kk, int buf) {
    const int k0 = kk * 32;
#pragma unroll
    for (int j = 0; j < 2; ++j) {
      const int chunk = tid + j * 256;
      const int ra = chunk >> 2, c = chunk & 3;
      async_cp16(&sA[buf][ra * LDA + c * 4],
                 hbf + (size_t)(b * NN + u0 + ra) * (DD / 2) + (k0 >> 1) + c * 4);
    }
    const int rb = tid >> 3, c = tid & 7;
    async_cp16(&sB[buf][rb * LDB + c * 4],
               wt + (size_t)(r * DD + k0 + rb) * (DD / 2) + (e0 >> 1) + c * 4);
  };

  v8f acc[4] = {};
  const int K = DD / 32;                     // 16 K-steps
  issueTile(0, 0);
  for (int kk = 0; kk < K; ++kk) {
    const int cur = kk % 3;
    if (kk + 1 < K) {
      issueTile(kk + 1, (kk + 1) % 3);
      wait_asynccnt<3>();                    // tile kk complete; kk+1 in flight
    } else {
      wait_asynccnt<0>();
    }
    __syncthreads();
    const v16bf af =
        fragA(&sA[cur][(wv * 16 + (lane & 15)) * LDA + (lane >> 4) * 4]);
#pragma unroll
    for (int t = 0; t < 4; ++t) {
      const v16bf bf = fragB(&sB[cur][(lane >> 4) * 8 * LDB], t * 16 + (lane & 15));
      acc[t] = __builtin_amdgcn_wmma_f32_16x16x32_bf16(
          false, af, false, bf, (short)0, acc[t], false, false);
    }
  }
  // store bf16: C layout: lane 0-15 N=lane, VGPR g -> M=g (+8 for lanes 16-31)
  const int col0 = e0 + (lane & 15);
  const int row0 = u0 + wv * 16 + ((lane >> 4) << 3);
#pragma unroll
  for (int t = 0; t < 4; ++t)
#pragma unroll
    for (int g = 0; g < 8; ++g)
      hw[((size_t)(br * NN + row0 + g)) * DD + col0 + t * 16] =
          (uint16_t)f2bf1(acc[t][g]);
}

// ------------------------- 5) agg[b] = sum_r (adj_r/deg) @ hW_r ---------------
// K = R*N = 5120. B tile (hW) is an async DMA; A tile (adj) needs f32->bf16
// with 1/deg scaling so it stays on the VALU path, written into the
// not-in-use pipeline buffer.
__global__ __launch_bounds__(256) void k_gemm_agg(const float* __restrict__ adj,
                                                  const float* __restrict__ rdeg,
                                                  const uint32_t* __restrict__ hw,
                                                  float* __restrict__ agg) {
  __shared__ __align__(16) uint32_t sA[3][128 * LDA];
  __shared__ __align__(16) uint32_t sB[3][32 * LDB];
  const int e0 = blockIdx.x * 64;
  const int v0 = blockIdx.y * 128;
  const int b  = blockIdx.z;
  const int tid = threadIdx.x, lane = tid & 31, wv = tid >> 5;

  auto issueB = [&](int kk, int buf) {
    const int r  = kk >> 5;
    const int u0 = (kk & 31) << 5;
    const int rb = tid >> 3, c = tid & 7;
    async_cp16(&sB[buf][rb * LDB + c * 4],
               hw + (size_t)((b * RR + r) * NN + u0 + rb) * (DD / 2) +
                   (e0 >> 1) + c * 4);
  };
  auto loadA = [&](int kk, int buf) {
    const int r  = kk >> 5;
    const int u0 = (kk & 31) << 5;
#pragma unroll
    for (int i = 0; i < 4; ++i) {
      const int chunk = tid + i * 256;
      const int ra = chunk >> 3, c4 = chunk & 7;
      const size_t base =
          ((size_t)((b * RR + r) * NN + v0 + ra)) * NN + u0 + c4 * 4;
      const float4 v = *reinterpret_cast<const float4*>(adj + base);
      __builtin_prefetch(adj + base + 32, 0, 0);   // stream next K slab
      const float s = rdeg[(b * RR + r) * NN + v0 + ra];
      sA[buf][ra * LDA + c4 * 2]     = pack2bf(v.x * s, v.y * s);
      sA[buf][ra * LDA + c4 * 2 + 1] = pack2bf(v.z * s, v.w * s);
    }
  };

  v8f acc[4] = {};
  const int K = RR * (NN / 32);              // 160 K-steps
  issueB(0, 0);
  loadA(0, 0);
  for (int kk = 0; kk < K; ++kk) {
    const int cur = kk % 3;
    if (kk + 1 < K) {
      issueB(kk + 1, (kk + 1) % 3);
      loadA(kk + 1, (kk + 1) % 3);
      wait_asynccnt<1>();                    // B tile kk complete
    } else {
      wait_asynccnt<0>();
    }
    __syncthreads();                         // also waits DScnt for A stores
    const v16bf af =
        fragA(&sA[cur][(wv * 16 + (lane & 15)) * LDA + (lane >> 4) * 4]);
#pragma unroll
    for (int t = 0; t < 4; ++t) {
      const v16bf bf = fragB(&sB[cur][(lane >> 4) * 8 * LDB], t * 16 + (lane & 15));
      acc[t] = __builtin_amdgcn_wmma_f32_16x16x32_bf16(
          false, af, false, bf, (short)0, acc[t], false, false);
    }
  }
  const int col0 = e0 + (lane & 15);
  const int row0 = v0 + wv * 16 + ((lane >> 4) << 3);
#pragma unroll
  for (int t = 0; t < 4; ++t)
#pragma unroll
    for (int g = 0; g < 8; ++g)
      agg[((size_t)(b * NN + row0 + g)) * DD + col0 + t * 16] = acc[t][g];
}

// ------------------------- 6) relu + residual + LayerNorm ---------------------
__global__ __launch_bounds__(256) void k_ln(const float* __restrict__ agg,
                                            const float* __restrict__ h,
                                            const float* __restrict__ gamma,
                                            const float* __restrict__ beta,
                                            float* __restrict__ out) {
  __shared__ float s1[256], s2[256];
  const int row = blockIdx.x;           // b*N + v
  const int tid = threadIdx.x;
  const size_t base = (size_t)row * DD;
  float x0 = fmaxf(agg[base + tid], 0.f) + h[base + tid];
  float x1 = fmaxf(agg[base + tid + 256], 0.f) + h[base + tid + 256];
  s1[tid] = x0 + x1;
  s2[tid] = x0 * x0 + x1 * x1;
  __syncthreads();
  for (int off = 128; off > 0; off >>= 1) {
    if (tid < off) { s1[tid] += s1[tid + off]; s2[tid] += s2[tid + off]; }
    __syncthreads();
  }
  const float mu  = s1[0] * (1.0f / DD);
  const float var = s2[0] * (1.0f / DD) - mu * mu;
  const float rs  = rsqrtf(var + LN_EPS);
  out[base + tid]       = (x0 - mu) * rs * gamma[tid] + beta[tid];
  out[base + tid + 256] = (x1 - mu) * rs * gamma[tid + 256] + beta[tid + 256];
}

// ------------------------- launch -------------------------
extern "C" void kernel_launch(void* const* d_in, const int* in_sizes, int n_in,
                              void* d_out, int out_size, void* d_ws, size_t ws_size,
                              hipStream_t stream) {
  const float* h     = (const float*)d_in[0];   // (B,N,D)
  const float* adj   = (const float*)d_in[1];   // (B,R,N,N)
  const float* W     = (const float*)d_in[2];   // (R,D,D)
  const float* gamma = (const float*)d_in[3];   // (D)
  const float* beta  = (const float*)d_in[4];   // (D)
  float* out = (float*)d_out;                   // (B,N,D)

  // Workspace layout (~67 MB total), all sections 256B-aligned.
  char* ws = (char*)d_ws;
  size_t o = 0;
  float*    rdeg = (float*)(ws + o);    o += (size_t)BB * RR * NN * sizeof(float); // 160 KB
  uint32_t* hbf  = (uint32_t*)(ws + o); o += (size_t)BB * NN * DD * 2;             // 8 MB
  uint16_t* wt   = (uint16_t*)(ws + o); o += (size_t)RR * DD * DD * 2;             // 2.5 MB
  uint16_t* hw   = (uint16_t*)(ws + o); o += (size_t)BB * RR * NN * DD * 2;        // 40 MB
  float*    agg  = (float*)(ws + o);    // 16 MB

  k_deg   <<<BB * RR * NN, 256, 0, stream>>>(adj, rdeg);
  k_cvt_h <<<(BB * NN * DD / 2) / 256, 256, 0, stream>>>(h, hbf);
  k_cvt_wt<<<(RR * DD * DD) / 256, 256, 0, stream>>>(W, wt);
  k_gemm_hw <<<dim3(DD / 64, NN / 128, BB * RR), 256, 0, stream>>>(
      hbf, (const uint32_t*)wt, hw);
  k_gemm_agg<<<dim3(DD / 64, NN / 128, BB), 256, 0, stream>>>(
      adj, rdeg, (const uint32_t*)hw, agg);
  k_ln<<<BB * NN, 256, 0, stream>>>(agg, h, gamma, beta, out);
}
